// LSTM_34256659153462
// MI455X (gfx1250) — matrix operand order repaired and checked by reading, not compiled, not run on hardware
//
#include <hip/hip_runtime.h>
#include <hip/hip_bf16.h>

// ---------------- problem constants ----------------
#define VOCAB 50000
#define DIM   300
#define HID   100
#define SEQT  512
#define BATCH 256
#define NOUT  3
#define G4    400          // 4*HID
#define KP0   320          // DIM padded to mult of 32
#define KPH   128          // HID padded to mult of 32
#define MROWS (BATCH*SEQT) // 131072
#define MTILES (MROWS/16)  // 8192

// padded LDS strides (avoid 64-dword-multiple rows -> bank conflicts)
#define WHHS  136          // W_hh LDS row stride in bf16 (272 B = 17 * 16 B)
#define HBS   136          // h LDS row stride in bf16
#define GS    404          // gate LDS row stride in f32

// ---------------- vector types ----------------
typedef __attribute__((ext_vector_type(16))) __bf16 v16bf;
typedef __attribute__((ext_vector_type(8)))  __bf16 v8bf;
typedef __attribute__((ext_vector_type(8)))  float  v8f;
typedef __attribute__((ext_vector_type(4)))  unsigned int u32x4;
typedef __attribute__((ext_vector_type(8)))  int i32x8;
typedef __attribute__((ext_vector_type(4)))  int i32x4;

union FragU { v16bf v; v8bf h[2]; };

__device__ __forceinline__ v8f wmma_bf16(v16bf a, v16bf b, v8f c) {
  // (neg_a, A, neg_b, B, c_mod, C, reuse_a, reuse_b)
  return __builtin_amdgcn_wmma_f32_16x16x32_bf16(false, a, false, b, (short)0, c,
                                                 false, false);
}

__device__ __forceinline__ float fast_sigmoid(float x) {
  x = fminf(fmaxf(x, -30.f), 30.f);
  return 1.f / (1.f + __expf(-x));
}
__device__ __forceinline__ float fast_tanh(float x) {
  x = fminf(fmaxf(x, -15.f), 15.f);
  float e = __expf(2.f * x);
  return (e - 1.f) / (e + 1.f);
}

// ---------------- workspace layout (bytes) ----------------
#define OFF_XBF  0ul
#define SZ_XBF   ((unsigned long)MROWS * KP0 * 2)        // 83,886,080  (reused as [*,KPH] for layers 1/2)
#define OFF_XG   (OFF_XBF + SZ_XBF)
#define SZ_XG    ((unsigned long)MROWS * G4 * 2)         // 104,857,600
#define OFF_W    (OFF_XG + SZ_XG)
// bf16-element offsets inside W region:
#define WIH0_E   0
#define WIH1_E   128000   // 400*320
#define WIH2_E   179200
#define WHH0_E   230400
#define WHH1_E   281600
#define WHH2_E   332800
#define WTOT_E   384000
#define OFF_BIAS (OFF_W + (unsigned long)WTOT_E * 2)
#define OFF_HLAST (OFF_BIAS + 1200ul * 4)

// =====================================================================
// Kernel 1: convert all weights to padded bf16, sum biases (b_ih + b_hh)
// =====================================================================
__global__ void lstm_wconv(const float* __restrict__ wih0, const float* __restrict__ whh0,
                           const float* __restrict__ bih0, const float* __restrict__ bhh0,
                           const float* __restrict__ wih1, const float* __restrict__ whh1,
                           const float* __restrict__ bih1, const float* __restrict__ bhh1,
                           const float* __restrict__ wih2, const float* __restrict__ whh2,
                           const float* __restrict__ bih2, const float* __restrict__ bhh2,
                           __bf16* __restrict__ Wbf, float* __restrict__ biassum) {
  int idx = blockIdx.x * blockDim.x + threadIdx.x;
  if (idx < WIH1_E) {                       // w_ih0: [400][300] -> [400][320]
    int row = idx / KP0, k = idx % KP0;
    float v = (k < DIM) ? wih0[row * DIM + k] : 0.f;
    Wbf[idx] = (__bf16)v;
  } else if (idx < WTOT_E) {                // five [400][100] -> [400][128] blocks
    int rel = idx - WIH1_E;
    int blk = rel / 51200;                  // 0:wih1 1:wih2 2:whh0 3:whh1 4:whh2
    int e   = rel % 51200;
    int row = e / KPH, k = e % KPH;
    const float* src = (blk == 0) ? wih1 : (blk == 1) ? wih2
                     : (blk == 2) ? whh0 : (blk == 3) ? whh1 : whh2;
    float v = (k < HID) ? src[row * HID + k] : 0.f;
    Wbf[idx] = (__bf16)v;
  } else if (idx < WTOT_E + 1200) {         // bias sums: 3 layers x 400
    int b = idx - WTOT_E;
    int l = b / G4, g = b % G4;
    const float* bi = (l == 0) ? bih0 : (l == 1) ? bih1 : bih2;
    const float* bh = (l == 0) ? bhh0 : (l == 1) ? bhh1 : bhh2;
    biassum[b] = bi[g] + bh[g];
  }
}

// =====================================================================
// Kernel 2: embedding gather -> bf16, padded [B*T][320]
// =====================================================================
__global__ void lstm_embed(const int* __restrict__ x, const float* __restrict__ emb,
                           __bf16* __restrict__ Xbf) {
  long r = blockIdx.x;                      // 0 .. B*T-1 ; row = b*T + t
  int tok = x[r];
  tok = (tok < 0) ? 0 : (tok >= VOCAB ? VOCAB - 1 : tok);
  const float* e = emb + (long)tok * DIM;
  __bf16* o = Xbf + r * KP0;
  for (int d = threadIdx.x; d < KP0; d += blockDim.x)
    o[d] = (d < DIM) ? (__bf16)e[d] : (__bf16)0.f;
}

// =====================================================================
// Kernel 3: WMMA GEMM  XG[M][400] = A[M][lda] @ W[400][lda]^T + bias
// One wave per 16-row M-tile; 5 groups of 5 N-tiles held in accumulators.
// =====================================================================
__global__ __launch_bounds__(256) void lstm_gemm(const __bf16* __restrict__ A, int lda,
                                                 int KT, const __bf16* __restrict__ W,
                                                 const float* __restrict__ bias,
                                                 __bf16* __restrict__ XG) {
  int lane = threadIdx.x & 31;
  int wid  = threadIdx.x >> 5;
  long mt = (long)blockIdx.x * 8 + wid;     // 8192 M-tiles
  if (mt >= MTILES) return;
  long mbase = mt * 16;
  int arow = lane & 15;                     // row / col within tile
  int asel = lane >> 4;

  for (int ng = 0; ng < 5; ++ng) {
    v8f z = {};
    v8f acc[5] = {z, z, z, z, z};
    for (int kt = 0; kt < KT; ++kt) {
      // A fragment 16x32 (row-major): lane<16 -> K 0..15, lane>=16 -> K 16..31 (contiguous)
      FragU au;
      const __bf16* ap = A + (mbase + arow) * (long)lda + kt * 32 + asel * 16;
      au.h[0] = *(const v8bf*)ap;
      au.h[1] = *(const v8bf*)(ap + 8);
      __builtin_prefetch((const void*)(ap + 32), 0, 1);   // next k-tile -> global_prefetch
      for (int i = 0; i < 5; ++i) {
        int nt = ng * 5 + i;
        // B fragment 32x16 from W[N][K]: lane<16 -> K 0..15, lane>=16 -> K 16..31 (contiguous)
        FragU bu;
        const __bf16* wp = W + (nt * 16 + arow) * (long)lda + kt * 32 + asel * 16;
        bu.h[0] = *(const v8bf*)wp;
        bu.h[1] = *(const v8bf*)(wp + 8);
        acc[i] = wmma_bf16(au.v, bu.v, acc[i]);
      }
    }
    // epilogue: C layout VGPR v -> M = v (+8 for lanes>=16), N = lane&15
    int rbase = (lane < 16) ? 0 : 8;
    for (int i = 0; i < 5; ++i) {
      int ncol = (ng * 5 + i) * 16 + arow;
      float bv = bias[ncol];
      for (int v = 0; v < 8; ++v) {
        long r = mbase + rbase + v;
        XG[r * G4 + ncol] = (__bf16)(acc[i][v] + bv);
      }
    }
  }
}

// =====================================================================
// Kernel 4: recurrence. grid = 16 tiles (16 batch rows each), 160 thr = 5 waves.
// LDS: W_hh bf16 [400][136] (TDM-loaded with row padding), h bf16 [16][136],
//      gate buffer f32 [16][404]. Each wave owns 5 N-tiles of h @ W_hh^T.
// =====================================================================
#define SM_WHH  (400 * WHHS * 2)                 // 108800
#define SM_HBF  (16 * HBS * 2)                   // 4352
#define SM_G    (16 * GS * 4)                    // 25856
#define REC_SMEM (SM_WHH + SM_HBF + SM_G)        // 139008

__global__ __launch_bounds__(160) void lstm_rec(const __bf16* __restrict__ xg,   // [B*T][400]
                                                const __bf16* __restrict__ whh,  // [400][128] packed
                                                __bf16* __restrict__ xnext,      // [B*T][128] or null
                                                float* __restrict__ hlast,       // [B][100] or null
                                                const int* __restrict__ lengths) {
  extern __shared__ char smem[];
  __bf16* s_whh = (__bf16*)smem;                       // [400][WHHS], at dynamic-LDS offset 0
  __bf16* s_hbf = (__bf16*)(smem + SM_WHH);            // [16][HBS]
  float*  s_g   = (float*)(smem + SM_WHH + SM_HBF);    // [16][GS]

  int tid  = threadIdx.x;
  int lane = tid & 31;
  int wid  = tid >> 5;
  int tile = blockIdx.x;

  // ---- stage W_hh into LDS via the Tensor Data Mover ----
  // D# describes a 1-D tile of 25600 x 8-byte elements; LDS padding inserts
  // 4 dwords after every 64 dwords (one 256B row) -> [400][136] bf16 layout.
#if defined(__gfx1250__)
  if (wid == 0) {
    unsigned long gaddr = (unsigned long)whh;
    u32x4 g0;
    g0.x = 1u;                                   // count=1 (valid user descriptor)
    g0.y = 0u;                                   // lds_addr = 0 (s_whh at dynamic-LDS base)
    g0.z = (unsigned)(gaddr & 0xffffffffu);      // global_addr[31:0]
    g0.w = (unsigned)(gaddr >> 32) | (2u << 30); // global_addr[56:32] | type=2 (image)
    i32x8 g1;
    g1[0] = (3 << 16)        // data_size = 8 bytes
          | (1 << 20)        // pad_enable
          | (5 << 22)        // pad_interval code 5 = every 64 dwords (one row)
          | (3 << 25);       // pad_amount code 3 = 4 dwords (16 B)
    g1[1] = (int)(25600u << 16);   // tensor_dim0[15:0] at [63:48]
    g1[2] = 0;                     // tensor_dim0 hi, tensor_dim1 lo
    g1[3] = (int)(25600u << 16);   // tile_dim0 = 25600 at [127:112]
    g1[4] = 0;                     // tile_dim1/tile_dim2 unused
    g1[5] = 25600;                 // tensor_dim0_stride lo
    g1[6] = 0;
    g1[7] = 0;
    i32x4 z4 = {};
    i32x8 z8 = {};
#if __clang_major__ >= 23
    __builtin_amdgcn_tensor_load_to_lds(g0, g1, z4, z4, z8, 0);
#else
    __builtin_amdgcn_tensor_load_to_lds(g0, g1, z4, z4, 0);
#endif
    __builtin_amdgcn_s_wait_tensorcnt(0);
    (void)z8;
  }
#else
  { // host-parse / non-gfx1250 fallback: padded copy, 16B chunks
    const uint4* src = (const uint4*)whh;
    for (int i = tid; i < 400 * 16; i += 160) {
      int row = i / 16, c = i % 16;
      *(uint4*)((char*)s_whh + row * (WHHS * 2) + c * 16) = src[row * 16 + c];
    }
  }
#endif
  for (int i = tid; i < 16 * HBS; i += 160) s_hbf[i] = (__bf16)0.f;

  // per-thread cell/hidden state: 10 (row,unit) pairs each
  float hreg[10], creg[10];
  int rowu[10], ju[10], lenu[10];
  for (int k = 0; k < 10; ++k) {
    int u = tid + 160 * k;                  // 0..1599
    rowu[k] = u / HID;
    ju[k]   = u % HID;
    hreg[k] = 0.f;
    creg[k] = 0.f;
    lenu[k] = lengths[tile * 16 + rowu[k]];
  }
  __syncthreads();

  int arow = lane & 15;
  int asel = lane >> 4;
  int rbase = (lane < 16) ? 0 : 8;

  for (int t = 0; t < SEQT; ++t) {
    // ---- preload this step's xg into registers (independent of WMMA, hides
    //      global latency behind the ds-load + WMMA phase) ----
    float xv[5][8];
    for (int i = 0; i < 5; ++i) {
      int ncol = (wid * 5 + i) * 16 + arow;
      long base = ((long)(tile * 16 + rbase) * SEQT + t) * G4 + ncol;
      for (int v = 0; v < 8; ++v)
        xv[i][v] = (float)xg[base + (long)v * SEQT * G4];
      // prefetch next timestep's lines (row stride is G4 elements)
      __builtin_prefetch((const void*)(xg + base + G4), 0, 0);
      __builtin_prefetch((const void*)(xg + base + 4 * (long)SEQT * G4 + G4), 0, 0);
    }

    // ---- WMMA phase: g = h @ W_hh^T (K = 128, padded h cols are zero) ----
    v8f z = {};
    v8f acc[5] = {z, z, z, z, z};
    for (int kt = 0; kt < 4; ++kt) {
      FragU au;
      const __bf16* ap = s_hbf + arow * HBS + kt * 32 + asel * 16;
      au.h[0] = *(const v8bf*)ap;
      au.h[1] = *(const v8bf*)(ap + 8);
      for (int i = 0; i < 5; ++i) {
        int nt = wid * 5 + i;
        FragU bu;
        const __bf16* wp = s_whh + (nt * 16 + arow) * WHHS + kt * 32 + asel * 16;
        bu.h[0] = *(const v8bf*)wp;
        bu.h[1] = *(const v8bf*)(wp + 8);
        acc[i] = wmma_bf16(au.v, bu.v, acc[i]);
      }
    }
    // epilogue: add xg, write pre-activation gates to LDS
    for (int i = 0; i < 5; ++i) {
      int ncol = (wid * 5 + i) * 16 + arow;
      for (int v = 0; v < 8; ++v) {
        int row = rbase + v;
        s_g[row * GS + ncol] = acc[i][v] + xv[i][v];
      }
    }
    __syncthreads();

    // ---- gate/activation phase (torch gate order i,f,g,o) ----
    for (int k = 0; k < 10; ++k) {
      int row = rowu[k], j = ju[k];
      float gi = s_g[row * GS + j];
      float gf = s_g[row * GS + 100 + j];
      float gg = s_g[row * GS + 200 + j];
      float go = s_g[row * GS + 300 + j];
      float ii = fast_sigmoid(gi);
      float ff = fast_sigmoid(gf);
      float gt = fast_tanh(gg);
      float oo = fast_sigmoid(go);
      float cn = ff * creg[k] + ii * gt;
      float hn = oo * fast_tanh(cn);
      bool m = (t < lenu[k]);               // hold state past sequence end
      creg[k] = m ? cn : creg[k];
      hreg[k] = m ? hn : hreg[k];
      s_hbf[row * HBS + j] = (__bf16)hreg[k];
      if (xnext) {
        long r = (long)(tile * 16 + row) * SEQT + t;
        xnext[r * KPH + j] = (__bf16)hreg[k];
      }
    }
    if (xnext) {                            // zero the K-padding of next-layer input
      for (int p = tid; p < 16 * (KPH - HID); p += 160) {
        int row = p / (KPH - HID);
        int j   = HID + p % (KPH - HID);
        long r = (long)(tile * 16 + row) * SEQT + t;
        xnext[r * KPH + j] = (__bf16)0.f;
      }
    }
    __syncthreads();
  }

  if (hlast)
    for (int k = 0; k < 10; ++k)
      hlast[(tile * 16 + rowu[k]) * HID + ju[k]] = hreg[k];
}

// =====================================================================
// Kernel 5: final linear head  out[B][3] = h_last @ w_fc^T + b_fc
// =====================================================================
__global__ void lstm_head(const float* __restrict__ hlast, const float* __restrict__ wfc,
                          const float* __restrict__ bfc, float* __restrict__ out) {
  int idx = blockIdx.x * blockDim.x + threadIdx.x;
  if (idx >= BATCH * NOUT) return;
  int b = idx / NOUT, o = idx % NOUT;
  float s = bfc[o];
  const float* hp = hlast + b * HID;
  const float* wp = wfc + o * HID;
  for (int j = 0; j < HID; ++j) s += hp[j] * wp[j];
  out[idx] = s;
}

// =====================================================================
extern "C" void kernel_launch(void* const* d_in, const int* in_sizes, int n_in,
                              void* d_out, int out_size, void* d_ws, size_t ws_size,
                              hipStream_t stream) {
  const int*   x       = (const int*)d_in[0];
  const int*   lengths = (const int*)d_in[1];
  const float* emb     = (const float*)d_in[2];
  const float* w_ih0 = (const float*)d_in[3];
  const float* w_hh0 = (const float*)d_in[4];
  const float* b_ih0 = (const float*)d_in[5];
  const float* b_hh0 = (const float*)d_in[6];
  const float* w_ih1 = (const float*)d_in[7];
  const float* w_hh1 = (const float*)d_in[8];
  const float* b_ih1 = (const float*)d_in[9];
  const float* b_hh1 = (const float*)d_in[10];
  const float* w_ih2 = (const float*)d_in[11];
  const float* w_hh2 = (const float*)d_in[12];
  const float* b_ih2 = (const float*)d_in[13];
  const float* b_hh2 = (const float*)d_in[14];
  const float* w_fc  = (const float*)d_in[15];
  const float* b_fc  = (const float*)d_in[16];

  char* ws = (char*)d_ws;
  __bf16* Xbf   = (__bf16*)(ws + OFF_XBF);
  __bf16* XG    = (__bf16*)(ws + OFF_XG);
  __bf16* Wbf   = (__bf16*)(ws + OFF_W);
  float*  biasS = (float*)(ws + OFF_BIAS);
  float*  hlast = (float*)(ws + OFF_HLAST);

  // 1) weight/bias conversion to padded bf16
  {
    int tot = WTOT_E + 1200;
    lstm_wconv<<<(tot + 255) / 256, 256, 0, stream>>>(
        w_ih0, w_hh0, b_ih0, b_hh0, w_ih1, w_hh1, b_ih1, b_hh1,
        w_ih2, w_hh2, b_ih2, b_hh2, Wbf, biasS);
  }
  // 2) embedding gather -> bf16 [B*T][320]
  lstm_embed<<<MROWS, 128, 0, stream>>>(x, emb, Xbf);

  // layer 0
  lstm_gemm<<<MTILES / 8, 256, 0, stream>>>(Xbf, KP0, KP0 / 32, Wbf + WIH0_E, biasS, XG);
  lstm_rec<<<16, 160, REC_SMEM, stream>>>(XG, Wbf + WHH0_E, Xbf, nullptr, lengths);
  // layer 1 (input now [B*T][128] bf16 in Xbf)
  lstm_gemm<<<MTILES / 8, 256, 0, stream>>>(Xbf, KPH, KPH / 32, Wbf + WIH1_E, biasS + G4, XG);
  lstm_rec<<<16, 160, REC_SMEM, stream>>>(XG, Wbf + WHH1_E, Xbf, nullptr, lengths);
  // layer 2
  lstm_gemm<<<MTILES / 8, 256, 0, stream>>>(Xbf, KPH, KPH / 32, Wbf + WIH2_E, biasS + 2 * G4, XG);
  lstm_rec<<<16, 160, REC_SMEM, stream>>>(XG, Wbf + WHH2_E, nullptr, hlast, lengths);

  // head
  lstm_head<<<(BATCH * NOUT + 255) / 256, 256, 0, stream>>>(hlast, w_fc, b_fc, (float*)d_out);
}
